// G_ATT_57724360458997
// MI455X (gfx1250) — matrix operand order repaired and checked by reading, not compiled, not run on hardware
//
#include <hip/hip_runtime.h>

// Problem constants (match reference setup_inputs)
#define B_  128
#define S_  128
#define E_  512
#define MT  16          // rows per workgroup tile
#define PF  520         // f32 LDS row pitch = 512 + 2 pads of 4 DWORDs (TDM pad layout)
#define PH  528         // bf16 LDS pitch

// bufA padded index: TDM inserts 4 DWORDs of pad after every 256 DWORDs stored,
// so column c of row r lives at r*520 + c + 4*(c>>8).
#define AIDX(r, c) ((r) * PF + (c) + ((((c) >> 8)) << 2))

typedef __bf16 v16bf __attribute__((ext_vector_type(16)));
typedef float  v8f   __attribute__((ext_vector_type(8)));

#if defined(__has_builtin)
#if __has_builtin(__builtin_amdgcn_tensor_load_to_lds) && \
    __has_builtin(__builtin_amdgcn_s_wait_tensorcnt)
#define HAVE_TDM 1
#endif
#endif
#ifndef HAVE_TDM
#define HAVE_TDM 0
#endif

#define STR2(x) #x
#define STR(x) STR2(x)
#if HAVE_TDM
#if __clang_major__ >= 23
#pragma message("CDNA5 MARKER: TDM path ENABLED (6-arg builtin), clang=" STR(__clang_major__))
#else
#pragma message("CDNA5 MARKER: TDM path ENABLED (5-arg builtin), clang=" STR(__clang_major__))
#endif
#else
#pragma message("CDNA5 MARKER: TDM path DISABLED (builtin missing), clang=" STR(__clang_major__))
#endif

#if HAVE_TDM
typedef unsigned u32x4 __attribute__((ext_vector_type(4)));
typedef int      i32x4 __attribute__((ext_vector_type(4)));
typedef int      i32x8 __attribute__((ext_vector_type(8)));
#endif

// ds_swizzle xor-reduce helpers (wave32-safe, no wavefront-size macro needed)
#define SWZ_F(v, imm) __int_as_float(__builtin_amdgcn_ds_swizzle(__float_as_int(v), (imm)))
#define XOR1  0x041F   // xor_mask=1 <<10 | and=0x1f
#define XOR2  0x081F
#define XOR4  0x101F
#define XOR8  0x201F

__global__ __launch_bounds__(256) void wconv_kernel(const float* __restrict__ w1,
                                                    const float* __restrict__ w0,
                                                    __bf16* __restrict__ o1,
                                                    __bf16* __restrict__ o0) {
  const int n = E_ * E_;
  for (int i = blockIdx.x * blockDim.x + threadIdx.x; i < n; i += gridDim.x * blockDim.x) {
    o1[i] = (__bf16)w1[i];
    o0[i] = (__bf16)w0[i];
  }
}

// t[b][o] = dot(C_target[b,0,:], Wg2[o,:]) + bias[o]
__global__ __launch_bounds__(256) void tvec_kernel(const float* __restrict__ Ct,
                                                   const float* __restrict__ Wg2,
                                                   const float* __restrict__ bias,
                                                   float* __restrict__ t) {
  const int b = blockIdx.x;
  const float4* ct = (const float4*)(Ct + (size_t)b * E_);
  for (int o = threadIdx.x; o < E_; o += blockDim.x) {
    const float4* w = (const float4*)(Wg2 + (size_t)o * E_);
    float s = 0.f;
#pragma unroll 4
    for (int i = 0; i < E_ / 4; ++i) {
      float4 a = ct[i], c = w[i];
      s += a.x * c.x + a.y * c.y + a.z * c.z + a.w * c.w;
    }
    t[(size_t)b * E_ + o] = s + bias[o];
  }
}

__global__ __launch_bounds__(256) void fused_entmax_attn(
    const float* __restrict__ C, const float* __restrict__ X,
    const __bf16* __restrict__ W1b, const __bf16* __restrict__ W0b,
    const float* __restrict__ tvec, const int* __restrict__ alpha_p,
    float* __restrict__ out) {
  __shared__ float  bufA[MT * PF];   // C tile (f32), later: score (f32)
  __shared__ __bf16 bufB[MT * PH];   // h (bf16), later: p (bf16)

  const int b    = blockIdx.x >> 3;          // S_/MT == 8 row tiles per batch
  const int m0   = (blockIdx.x & 7) * MT;
  const int tid  = threadIdx.x;
  const int wv   = tid >> 5;
  const int lane = tid & 31;
  const int hm   = lane & 15;                // A row / B,D column within tile
  const int kh   = lane >> 4;                // lane-half selects K split / D row-half
  const int ncol0 = wv * 64 + hm;            // this wave owns columns [wv*64, wv*64+64)
  const int drow  = kh * 8;

  const float a1     = (float)(*alpha_p) - 1.0f;
  const float inv_a1 = 1.0f / a1;

  // ---- stage 0: C[b, m0:m0+16, :] fp32 -> LDS via Tensor Data Mover ----
#if HAVE_TDM
  if (wv == 0) {
    const unsigned lds_off = (unsigned)(size_t)(&bufA[0]);
    const unsigned long long ga =
        (unsigned long long)(size_t)(C + (size_t)b * S_ * E_ + (size_t)m0 * E_);
    u32x4 g0;
    g0[0] = 1u;                                   // count=1 user descriptor
    g0[1] = lds_off;                              // lds_addr (bytes)
    g0[2] = (unsigned)ga;                         // global_addr[31:0]
    g0[3] = (unsigned)((ga >> 32) & 0x01FFFFFFull) | (2u << 30);  // addr[56:32], type=2
    i32x8 g1;
    g1[0] = (int)((2u << 16) |                    // data_size = 4 bytes
                  (1u << 20) |                    // pad_enable
                  (7u << 22) |                    // pad_interval: every 256 DWORDs
                  (3u << 25));                    // pad_amount: 4 DWORDs
    g1[1] = (int)(512u << 16);                    // tensor_dim0 = 512 (low 16)
    g1[2] = (int)((unsigned)MT << 16);            // tensor_dim1 = 16 (low 16)
    g1[3] = (int)(512u << 16);                    // tile_dim0 = 512
    g1[4] = MT;                                   // tile_dim1 = 16, tile_dim2 = 0
    g1[5] = 512;                                  // tensor_dim0_stride = 512
    g1[6] = 0;
    g1[7] = 0;
    i32x4 z4 = {0, 0, 0, 0};
#if __clang_major__ >= 23
    i32x8 z8 = {0, 0, 0, 0, 0, 0, 0, 0};
    __builtin_amdgcn_tensor_load_to_lds(g0, g1, z4, z4, z8, 0);
#else
    __builtin_amdgcn_tensor_load_to_lds(g0, g1, z4, z4, 0);
#endif
    __builtin_amdgcn_s_wait_tensorcnt(0);
  }
#else
  {
    const float4* src = (const float4*)(C + (size_t)b * S_ * E_ + (size_t)m0 * E_);
    for (int i = tid; i < MT * (E_ / 4); i += 256) {
      const int r = i >> 7, c4 = i & 127;           // 128 float4 per row
      *(float4*)&bufA[AIDX(r, c4 * 4)] = src[r * (E_ / 4) + c4];
    }
  }
#endif
  __syncthreads();

  v8f acc[4];

  // ---- stage 1: H = relu(C @ W1^T + t) ----
#pragma unroll
  for (int nt = 0; nt < 4; ++nt)
#pragma unroll
    for (int r = 0; r < 8; ++r) acc[nt][r] = 0.f;

  for (int k0 = 0; k0 < E_; k0 += 32) {
    v16bf a;
#pragma unroll
    for (int j = 0; j < 8; ++j) {                    // contiguous K order (k = 16*kh + s), matches B
      a[j]     = (__bf16)bufA[AIDX(hm, k0 + kh * 16 +     j)];
      a[j + 8] = (__bf16)bufA[AIDX(hm, k0 + kh * 16 + 8 + j)];
    }
#pragma unroll
    for (int nt = 0; nt < 4; ++nt) {
      const v16bf bm = *(const v16bf*)(W1b + (size_t)(ncol0 + nt * 16) * E_ + k0 + kh * 16);
      acc[nt] = __builtin_amdgcn_wmma_f32_16x16x32_bf16(false, a, false, bm,
                                                        (short)0, acc[nt], false, false);
    }
  }
#pragma unroll
  for (int nt = 0; nt < 4; ++nt) {
    const float tv = tvec[(size_t)b * E_ + ncol0 + nt * 16];
#pragma unroll
    for (int r = 0; r < 8; ++r) {
      float v = acc[nt][r] + tv;
      bufB[(drow + r) * PH + ncol0 + nt * 16] = (__bf16)fmaxf(v, 0.f);
    }
  }
  __syncthreads();

  // ---- stage 2: score = H @ W0^T  (f32 into bufA) ----
#pragma unroll
  for (int nt = 0; nt < 4; ++nt)
#pragma unroll
    for (int r = 0; r < 8; ++r) acc[nt][r] = 0.f;

  for (int k0 = 0; k0 < E_; k0 += 32) {
    v16bf a;
#pragma unroll
    for (int j = 0; j < 8; ++j) {
      a[j]     = bufB[hm * PH + k0 + kh * 16 +     j];
      a[j + 8] = bufB[hm * PH + k0 + kh * 16 + 8 + j];
    }
#pragma unroll
    for (int nt = 0; nt < 4; ++nt) {
      const v16bf bm = *(const v16bf*)(W0b + (size_t)(ncol0 + nt * 16) * E_ + k0 + kh * 16);
      acc[nt] = __builtin_amdgcn_wmma_f32_16x16x32_bf16(false, a, false, bm,
                                                        (short)0, acc[nt], false, false);
    }
  }
#pragma unroll
  for (int nt = 0; nt < 4; ++nt)
#pragma unroll
    for (int r = 0; r < 8; ++r)
      bufA[AIDX(drow + r, ncol0 + nt * 16)] = acc[nt][r];
  __syncthreads();

  // ---- stage 3: entmax bisection over each row of 512 (16 lanes per row) ----
  {
    const int row   = wv * 2 + kh;     // 8 waves * 2 rows = 16 rows
    const int cbase = hm * 32;         // each lane owns 32 contiguous features
    float xs[32];
#pragma unroll
    for (int i = 0; i < 32; ++i) xs[i] = bufA[AIDX(row, cbase + i)] * a1;

    float rmax = xs[0];
#pragma unroll
    for (int i = 1; i < 32; ++i) rmax = fmaxf(rmax, xs[i]);
    rmax = fmaxf(rmax, SWZ_F(rmax, XOR1));
    rmax = fmaxf(rmax, SWZ_F(rmax, XOR2));
    rmax = fmaxf(rmax, SWZ_F(rmax, XOR4));
    rmax = fmaxf(rmax, SWZ_F(rmax, XOR8));

    const bool lin = (inv_a1 == 1.0f);           // alpha == 2 fast path
    const float ghi = lin ? (1.0f / (float)E_) : __powf(1.0f / (float)E_, a1);
    float tau = rmax - 1.0f;
    float dm  = (rmax - ghi) - tau;
    for (int it = 0; it < 50; ++it) {
      dm *= 0.5f;
      const float tm = tau + dm;
      float s = 0.f;
#pragma unroll
      for (int i = 0; i < 32; ++i) {
        float v = fmaxf(xs[i] - tm, 0.f);
        s += lin ? v : __powf(v, inv_a1);
      }
      s += SWZ_F(s, XOR1);
      s += SWZ_F(s, XOR2);
      s += SWZ_F(s, XOR4);
      s += SWZ_F(s, XOR8);
      if (s - 1.0f >= 0.f) tau = tm;
    }
    float p[32], ssum = 0.f;
#pragma unroll
    for (int i = 0; i < 32; ++i) {
      float v = fmaxf(xs[i] - tau, 0.f);
      v = lin ? v : __powf(v, inv_a1);
      p[i] = v;
      ssum += v;
    }
    ssum += SWZ_F(ssum, XOR1);
    ssum += SWZ_F(ssum, XOR2);
    ssum += SWZ_F(ssum, XOR4);
    ssum += SWZ_F(ssum, XOR8);
    const float rs = 1.0f / ssum;
#pragma unroll
    for (int i = 0; i < 32; ++i)
      bufB[row * PH + cbase + i] = (__bf16)(p[i] * rs);
  }
  __syncthreads();

  // ---- stage 4: out = P @ X[b] ----
#pragma unroll
  for (int nt = 0; nt < 4; ++nt)
#pragma unroll
    for (int r = 0; r < 8; ++r) acc[nt][r] = 0.f;

  const float* Xb = X + (size_t)b * E_ * E_;
  for (int k0 = 0; k0 < E_; k0 += 32) {
    if (k0 + 32 < E_)
      __builtin_prefetch(&Xb[(size_t)(k0 + 32 + kh * 16) * E_ + ncol0], 0, 0);
    v16bf a;
#pragma unroll
    for (int j = 0; j < 8; ++j) {
      a[j]     = bufB[hm * PH + k0 + kh * 16 +     j];
      a[j + 8] = bufB[hm * PH + k0 + kh * 16 + 8 + j];
    }
#pragma unroll
    for (int nt = 0; nt < 4; ++nt) {
      v16bf bm;
#pragma unroll
      for (int j = 0; j < 16; ++j)     // B[k][n] = X[b][k][n], converted in-flight
        bm[j] = (__bf16)Xb[(size_t)(k0 + kh * 16 + j) * E_ + ncol0 + nt * 16];
      acc[nt] = __builtin_amdgcn_wmma_f32_16x16x32_bf16(false, a, false, bm,
                                                        (short)0, acc[nt], false, false);
    }
  }
  float* outp = out + (size_t)b * S_ * E_ + (size_t)m0 * E_;
#pragma unroll
  for (int nt = 0; nt < 4; ++nt)
#pragma unroll
    for (int r = 0; r < 8; ++r)
      outp[(size_t)(drow + r) * E_ + ncol0 + nt * 16] = acc[nt][r];
}

extern "C" void kernel_launch(void* const* d_in, const int* in_sizes, int n_in,
                              void* d_out, int out_size, void* d_ws, size_t ws_size,
                              hipStream_t stream) {
  const float* C_target = (const float*)d_in[0];
  const float* C        = (const float*)d_in[1];
  const float* X        = (const float*)d_in[2];
  const float* Wg1      = (const float*)d_in[3];
  const float* Wg2      = (const float*)d_in[4];
  const float* Wg0      = (const float*)d_in[5];
  const float* bias     = (const float*)d_in[6];
  const int*   alpha    = (const int*)d_in[7];

  char* ws = (char*)d_ws;
  __bf16* W1b = (__bf16*)ws;                                   // 512*512*2 B
  __bf16* W0b = (__bf16*)(ws + (size_t)E_ * E_ * 2);           // 512*512*2 B
  float*  tv  = (float*)(ws + (size_t)E_ * E_ * 4);            // 128*512*4 B

  wconv_kernel<<<256, 256, 0, stream>>>(Wg1, Wg0, W1b, W0b);
  tvec_kernel<<<B_, 256, 0, stream>>>(C_target, Wg2, bias, tv);
  fused_entmax_attn<<<B_ * (S_ / MT), 256, 0, stream>>>(C, X, W1b, W0b, tv, alpha,
                                                        (float*)d_out);
}
